// FocusNetwork_79422535238091
// MI455X (gfx1250) — compile-verified, matrix operand order checked by reading, not run on hardware
//
#include <hip/hip_runtime.h>
#include <hip/hip_bf16.h>

typedef __attribute__((ext_vector_type(16))) _Float16 v16h;
typedef __attribute__((ext_vector_type(8)))  _Float16 v8h;
typedef __attribute__((ext_vector_type(8)))  float    v8f;

#define DEV static __device__ __forceinline__

DEV int imin(int a, int b) { return a < b ? a : b; }
DEV int imax(int a, int b) { return a > b ? a : b; }

// f16 WMMA 16x16x32 fragment: per ISA, each lane holds two contiguous 8-element
// K runs. lanes 0-15: K {0..7} u {16..23}; lanes 16-31: K {8..15} u {24..31}.
// With K contiguous in memory this is two aligned 16B loads + static shuffle.
DEV v16h load_frag(const _Float16* __restrict__ base, int hi_half) {
  const _Float16* p = base + (hi_half ? 8 : 0);
  v8h lo = *(const v8h*)(p);        // K run 0
  v8h hi = *(const v8h*)(p + 16);   // K run 1
  return __builtin_shufflevector(lo, hi, 0, 1, 2, 3, 4, 5, 6, 7,
                                 8, 9, 10, 11, 12, 13, 14, 15);
}

// ---------------------------------------------------------------- sizes
constexpr int BATCH = 128;
constexpr int IH = 224, IW = 224;
constexpr int PR = 64;          // crop-resize output resolution
constexpr int C1 = 64;          // conv1 out channels
constexpr int C2 = 128;         // conv2 out channels
constexpr int MIP = 8;
constexpr int DM = 512;
constexpr int PAD = 34;         // 32 + zero halo for conv2

// ---------------------------------------------------------------- K1: crop + bilinear resize
__global__ void crop_resize_kernel(const float* __restrict__ img,
                                   const float* __restrict__ pos,
                                   float* __restrict__ patch) {
  int idx = blockIdx.x * blockDim.x + threadIdx.x;     // B*3*64*64
  if (idx >= BATCH * 3 * PR * PR) return;
  int j = idx & 63, i = (idx >> 6) & 63;
  int c = (idx >> 12) % 3;
  int b = idx / (3 * PR * PR);

  const int ps = 56;                                   // min(224,224)/4
  float px = pos[b * 2 + 0], py = pos[b * 2 + 1];
  int cx = (int)floorf(px * (float)IW);
  int cy = (int)floorf(py * (float)IH);
  int x1 = imax(0, cx - ps / 2);
  int x2 = imin(IW, x1 + ps);
  int y1 = imax(0, cy - ps / 2);
  int y2 = imin(IH, y1 + ps);
  int wi = x2 - x1, hi = y2 - y1;
  float w = (float)wi, h = (float)hi;

  float dx = (j + 0.5f) / (float)PR;
  float dy = (i + 0.5f) / (float)PR;
  float rx = fmaxf(dx * w - 0.5f, 0.f);
  float ry = fmaxf(dy * h - 0.5f, 0.f);
  float fx = rx - floorf(rx), fy = ry - floorf(ry);
  int xi0 = imin(imax((int)floorf(rx), 0), wi - 1);
  int xi1 = imin(xi0 + 1, wi - 1);
  int yi0 = imin(imax((int)floorf(ry), 0), hi - 1);
  int yi1 = imin(yi0 + 1, hi - 1);

  const float* ip = img + ((size_t)b * 3 + c) * IH * IW;
  float v00 = ip[(y1 + yi0) * IW + x1 + xi0];
  float v01 = ip[(y1 + yi0) * IW + x1 + xi1];
  float v10 = ip[(y1 + yi1) * IW + x1 + xi0];
  float v11 = ip[(y1 + yi1) * IW + x1 + xi1];
  float r0 = v00 * (1.f - fx) + v01 * fx;
  float r1 = v10 * (1.f - fx) + v11 * fx;
  patch[idx] = r0 * (1.f - fy) + r1 * fy;
}

// ---------------------------------------------------------------- K2: conv1 5x5 pad2, direct f32
__global__ void conv1_kernel(const float* __restrict__ patch,
                             const float* __restrict__ w,
                             const float* __restrict__ bias,
                             float* __restrict__ y1) {
  int idx = blockIdx.x * blockDim.x + threadIdx.x;     // B*64*64*64
  if (idx >= BATCH * C1 * PR * PR) return;
  int x = idx & 63, y = (idx >> 6) & 63;
  int oc = (idx >> 12) & 63;
  int b = idx >> 18;
  float acc = bias[oc];
  for (int c = 0; c < 3; ++c) {
    const float* ip = patch + ((size_t)b * 3 + c) * PR * PR;
    const float* wp = w + ((size_t)oc * 3 + c) * 25;
    for (int ky = 0; ky < 5; ++ky) {
      int iy = y + ky - 2;
      if (iy < 0 || iy >= PR) continue;
      for (int kx = 0; kx < 5; ++kx) {
        int ix = x + kx - 2;
        if (ix < 0 || ix >= PR) continue;
        acc = fmaf(ip[iy * PR + ix], wp[ky * 5 + kx], acc);
      }
    }
  }
  y1[idx] = acc;
}

// ---------------------------------------------------------------- K3: per-channel batch stats
__global__ void stats_kernel(const float* __restrict__ x,
                             float* __restrict__ meanv,
                             float* __restrict__ rstdv,
                             int Bn, int C, int HW) {
  int c = blockIdx.x;
  int tid = threadIdx.x;
  int N = Bn * HW;
  float s = 0.f, s2 = 0.f;
  for (int t = tid; t < N; t += blockDim.x) {
    int b = t / HW, i = t - b * HW;
    float v = x[((size_t)b * C + c) * HW + i];
    s += v;
    s2 += v * v;
  }
  __shared__ float sh[256], sh2[256];
  sh[tid] = s; sh2[tid] = s2;
  __syncthreads();
  for (int off = 128; off > 0; off >>= 1) {
    if (tid < off) { sh[tid] += sh[tid + off]; sh2[tid] += sh2[tid + off]; }
    __syncthreads();
  }
  if (tid == 0) {
    float m = sh[0] / (float)N;
    float var = sh2[0] / (float)N - m * m;
    meanv[c] = m;
    rstdv[c] = rsqrtf(var + 1e-5f);
  }
}

// ---------------------------------------------------------------- K4: bn1 + relu + 2x2 maxpool -> padded NHWC f16
__global__ void bn_relu_pool_kernel(const float* __restrict__ y1,
                                    const float* __restrict__ mean,
                                    const float* __restrict__ rstd,
                                    const float* __restrict__ g,
                                    const float* __restrict__ bb,
                                    _Float16* __restrict__ x1p) {
  int idx = blockIdx.x * blockDim.x + threadIdx.x;     // B*64*32*32
  if (idx >= BATCH * C1 * 32 * 32) return;
  int x = idx & 31, y = (idx >> 5) & 31;
  int c = (idx >> 10) & 63;
  int b = idx >> 16;
  const float* p = y1 + (((size_t)b * C1 + c) * PR + 2 * y) * PR + 2 * x;
  float v0 = p[0], v1 = p[1], v2 = p[PR], v3 = p[PR + 1];
  float mx = fmaxf(fmaxf(v0, v1), fmaxf(v2, v3));
  float nv = (mx - mean[c]) * rstd[c] * g[c] + bb[c];  // relu(max(bn)) == max(relu(bn))
  // NHWC with +1 halo so conv2 needs no bounds checks
  x1p[(((size_t)b * PAD + (y + 1)) * PAD + (x + 1)) * C1 + c] =
      (_Float16)fmaxf(nv, 0.f);
}

__global__ void zero_h_kernel(_Float16* __restrict__ p, int n) {
  int i = blockIdx.x * blockDim.x + threadIdx.x;
  if (i < n) p[i] = (_Float16)0.f;
}

// transpose conv2_w (O=128, I=64, 3, 3) f32 -> w2t[(ky*3+kx)][o][i] f16 (i fastest)
__global__ void cvt_w2_kernel(const float* __restrict__ src,
                              _Float16* __restrict__ dst) {
  int idx = blockIdx.x * blockDim.x + threadIdx.x;     // 9*128*64
  if (idx >= 9 * C2 * C1) return;
  int i = idx & 63;
  int o = (idx >> 6) & 127;
  int t = idx >> 13;                                   // tap = ky*3+kx
  int ky = t / 3, kx = t - ky * 3;
  dst[idx] = (_Float16)src[(((size_t)o * C1 + i) * 3 + ky) * 3 + kx];
}

// ---------------------------------------------------------------- K5: conv2 3x3 pad1 via WMMA f16
// Implicit GEMM: M = 16 out-channels, N = 16 pixels of one row, K = 64ch x 9 taps.
// Channel-contiguous layouts -> each fragment = two global_load_b128.
// Block = 512 threads = 16 waves; wave -> (mtile 0..7, xtile 0..1); blockIdx -> (b, y).
__global__ void conv2_wmma_kernel(const _Float16* __restrict__ x1p,
                                  const _Float16* __restrict__ w2t,
                                  const float* __restrict__ bias,
                                  float* __restrict__ y2) {
  int lane = threadIdx.x & 31;
  int wave = threadIdx.x >> 5;                          // 0..15
  int mtile = wave >> 1;                                // 0..7
  int xtile = wave & 1;
  int b = blockIdx.x >> 5;
  int y = blockIdx.x & 31;
  int hh = (lane >> 4) & 1;

  int m = mtile * 16 + (lane & 15);                     // out channel (A row)
  int n = xtile * 16 + (lane & 15);                     // pixel x (B col)

  v8f acc = {};
#pragma unroll
  for (int ky = 0; ky < 3; ++ky) {
#pragma unroll
    for (int kx = 0; kx < 3; ++kx) {
      const _Float16* wtap = w2t + (((size_t)(ky * 3 + kx) * C2 + m) * C1);
      const _Float16* itap =
          x1p + (((size_t)b * PAD + (y + ky)) * PAD + (n + kx)) * C1;
#pragma unroll
      for (int ch = 0; ch < 2; ++ch) {                  // K chunks of 32 channels
        v16h af = load_frag(wtap + ch * 32, hh);
        v16h bf = load_frag(itap + ch * 32, hh);
        acc = __builtin_amdgcn_wmma_f32_16x16x32_f16(
            false, af, false, bf, (short)0, acc, false, false);
      }
    }
  }
  // D layout: VGPR r -> lanes 0-15: M=r, N=lane; lanes 16-31: M=r+8, N=lane-16
  int nn = xtile * 16 + (lane & 15);
#pragma unroll
  for (int r = 0; r < 8; ++r) {
    int mm = r + ((lane & 16) ? 8 : 0);
    int oc = mtile * 16 + mm;
    y2[(((size_t)b * C2 + oc) * 32 + y) * 32 + nn] = acc[r] + bias[oc];
  }
}

// ---------------------------------------------------------------- K6: bn2 + relu -> f32
__global__ void bn2_relu_kernel(const float* __restrict__ y2,
                                const float* __restrict__ mean,
                                const float* __restrict__ rstd,
                                const float* __restrict__ g,
                                const float* __restrict__ bb,
                                float* __restrict__ x2) {
  int idx = blockIdx.x * blockDim.x + threadIdx.x;     // B*128*32*32
  if (idx >= BATCH * C2 * 32 * 32) return;
  int c = (idx >> 10) & 127;
  float v = (y2[idx] - mean[c]) * rstd[c] * g[c] + bb[c];
  x2[idx] = fmaxf(v, 0.f);
}

// ---------------------------------------------------------------- K7: coord-att H/W means -> ycat (B,128,64)
__global__ void ca_pool_kernel(const float* __restrict__ x2,
                               float* __restrict__ ycat) {
  int idx = blockIdx.x * blockDim.x + threadIdx.x;     // B*128*64
  if (idx >= BATCH * C2 * 64) return;
  int s = idx & 63;
  int c = (idx >> 6) & 127;
  int b = idx >> 13;
  const float* p = x2 + ((size_t)b * C2 + c) * 1024;
  float acc = 0.f;
  if (s < 32) {                                        // xh: mean over W of row s
    for (int w = 0; w < 32; ++w) acc += p[s * 32 + w];
  } else {                                             // xw: mean over H of col s-32
    int w = s - 32;
    for (int h = 0; h < 32; ++h) acc += p[h * 32 + w];
  }
  ycat[idx] = acc * (1.f / 32.f);
}

// ---------------------------------------------------------------- K8: ca 1x1 conv 128->8 (pre-BN)
__global__ void ca1_kernel(const float* __restrict__ ycat,
                           const float* __restrict__ w,
                           const float* __restrict__ bias,
                           float* __restrict__ ca1) {
  int idx = blockIdx.x * blockDim.x + threadIdx.x;     // B*8*64
  if (idx >= BATCH * MIP * 64) return;
  int s = idx & 63;
  int o = (idx >> 6) & 7;
  int b = idx >> 9;
  float acc = bias[o];
  const float* yp = ycat + (size_t)b * C2 * 64 + s;
  for (int i = 0; i < C2; ++i)
    acc = fmaf(yp[(size_t)i * 64], w[o * C2 + i], acc);
  ca1[idx] = acc;
}

// ---------------------------------------------------------------- K9: ca bn + relu (in place)
__global__ void ca_bnrelu_kernel(float* __restrict__ ca1,
                                 const float* __restrict__ mean,
                                 const float* __restrict__ rstd,
                                 const float* __restrict__ g,
                                 const float* __restrict__ bb) {
  int idx = blockIdx.x * blockDim.x + threadIdx.x;     // B*8*64
  if (idx >= BATCH * MIP * 64) return;
  int c = (idx >> 6) & 7;
  float v = (ca1[idx] - mean[c]) * rstd[c] * g[c] + bb[c];
  ca1[idx] = fmaxf(v, 0.f);
}

// ---------------------------------------------------------------- K10: attention maps (sigmoid 1x1 convs)
__global__ void ca_att_kernel(const float* __restrict__ ca1,
                              const float* __restrict__ chw, const float* __restrict__ chb,
                              const float* __restrict__ cww, const float* __restrict__ cwb,
                              float* __restrict__ ahw) {
  int idx = blockIdx.x * blockDim.x + threadIdx.x;     // B*128*64
  if (idx >= BATCH * C2 * 64) return;
  int s = idx & 63;
  int o = (idx >> 6) & 127;
  int b = idx >> 13;
  const float* yp = ca1 + (size_t)b * MIP * 64 + s;
  const float* wv = (s < 32) ? (chw + o * MIP) : (cww + o * MIP);
  float acc = (s < 32) ? chb[o] : cwb[o];
  for (int i = 0; i < MIP; ++i)
    acc = fmaf(yp[(size_t)i * 64], wv[i], acc);
  ahw[idx] = 1.f / (1.f + expf(-acc));
}

// ---------------------------------------------------------------- K11: apply attention + 4x4 avg pool -> f16 (B,2048)
__global__ void pool16_kernel(const float* __restrict__ x2,
                              const float* __restrict__ ahw,
                              _Float16* __restrict__ act) {
  int idx = blockIdx.x * blockDim.x + threadIdx.x;     // B*128*16
  if (idx >= BATCH * C2 * 16) return;
  int j = idx & 3, i = (idx >> 2) & 3;
  int c = (idx >> 4) & 127;
  int b = idx >> 11;
  const float* p = x2 + ((size_t)b * C2 + c) * 1024;
  const float* ah = ahw + ((size_t)b * C2 + c) * 64;
  const float* aw = ah + 32;
  float s = 0.f;
  for (int dh = 0; dh < 8; ++dh) {
    int h = i * 8 + dh;
    float fh = ah[h];
    for (int dw = 0; dw < 8; ++dw) {
      int w = j * 8 + dw;
      s = fmaf(p[h * 32 + w] * fh, aw[w], s);
    }
  }
  act[(size_t)b * 2048 + c * 16 + i * 4 + j] = (_Float16)(s * (1.f / 64.f));
}

// ---------------------------------------------------------------- K12: FC (128x2048)@(2048x512) via WMMA f16
__global__ void fc_wmma_kernel(const _Float16* __restrict__ act,
                               const _Float16* __restrict__ fcwh,
                               const float* __restrict__ fc_b,
                               float* __restrict__ out) {
  int lane = threadIdx.x & 31;
  int wave = threadIdx.x >> 5;                          // 0..7
  int gwave = blockIdx.x * 8 + wave;                    // 0..255
  int mtile = gwave >> 5;                               // 0..7
  int ntile = gwave & 31;                               // 0..31
  int hh = (lane >> 4) & 1;

  int mrow = mtile * 16 + (lane & 15);                  // batch row
  int ncol = ntile * 16 + (lane & 15);                  // out feature

  const _Float16* arow = act + (size_t)mrow * 2048;
  const _Float16* brow = fcwh + (size_t)ncol * 2048;    // B[k][n] = W[n][k]

  v8f acc = {};
  for (int kk = 0; kk < 64; ++kk) {
    // prefetch next K-chunk of activations (global_prefetch_b8)
    __builtin_prefetch(arow + imin(kk + 1, 63) * 32, 0, 0);
    v16h af = load_frag(arow + kk * 32, hh);
    v16h bf = load_frag(brow + kk * 32, hh);
    acc = __builtin_amdgcn_wmma_f32_16x16x32_f16(
        false, af, false, bf, (short)0, acc, false, false);
  }
  int nn = ntile * 16 + (lane & 15);
#pragma unroll
  for (int r = 0; r < 8; ++r) {
    int mm = r + ((lane & 16) ? 8 : 0);
    int bi = mtile * 16 + mm;
    out[(size_t)bi * DM + nn] = acc[r] + fc_b[nn];
  }
}

// ---------------------------------------------------------------- misc
__global__ void cvt_f16_kernel(const float* __restrict__ src,
                               _Float16* __restrict__ dst, int n) {
  int i = blockIdx.x * blockDim.x + threadIdx.x;
  if (i < n) dst[i] = (_Float16)src[i];
}

extern "C" void kernel_launch(void* const* d_in, const int* in_sizes, int n_in,
                              void* d_out, int out_size, void* d_ws, size_t ws_size,
                              hipStream_t stream) {
  (void)in_sizes; (void)n_in; (void)out_size; (void)ws_size;
  const float* image    = (const float*)d_in[0];
  const float* position = (const float*)d_in[1];
  const float* conv1_w  = (const float*)d_in[2];
  const float* conv1_b  = (const float*)d_in[3];
  const float* bn1_g    = (const float*)d_in[4];
  const float* bn1_b    = (const float*)d_in[5];
  const float* conv2_w  = (const float*)d_in[6];
  const float* conv2_b  = (const float*)d_in[7];
  const float* bn2_g    = (const float*)d_in[8];
  const float* bn2_b    = (const float*)d_in[9];
  const float* ca_c1_w  = (const float*)d_in[10];
  const float* ca_c1_b  = (const float*)d_in[11];
  const float* ca_bn_g  = (const float*)d_in[12];
  const float* ca_bn_b  = (const float*)d_in[13];
  const float* ca_ch_w  = (const float*)d_in[14];
  const float* ca_ch_b  = (const float*)d_in[15];
  const float* ca_cw_w  = (const float*)d_in[16];
  const float* ca_cw_b  = (const float*)d_in[17];
  const float* fc_w     = (const float*)d_in[18];
  const float* fc_b     = (const float*)d_in[19];
  float* out = (float*)d_out;

  char* ws = (char*)d_ws;
  size_t off = 0;
  auto alloc = [&](size_t bytes) -> char* {
    char* p = ws + off;
    off += (bytes + 255) & ~(size_t)255;
    return p;
  };

  const int NPAD = BATCH * PAD * PAD * C1;                                  // padded NHWC f16
  float*    patch  = (float*)alloc((size_t)BATCH * 3 * PR * PR * 4);        // 6.3 MB
  char*     region = alloc((size_t)BATCH * C1 * PR * PR * 4);               // 134 MB, reused
  float*    y1     = (float*)region;                                        // conv1 raw
  float*    y2     = (float*)region;                                        // conv2 raw (y1 dead)
  float*    x2     = (float*)(region + (size_t)BATCH * C2 * 1024 * 4);      // normalized conv2
  _Float16* x1p    = (_Float16*)alloc((size_t)NPAD * 2);                    // 18.9 MB
  _Float16* w2t    = (_Float16*)alloc((size_t)9 * C2 * C1 * 2);
  _Float16* fcwh   = (_Float16*)alloc((size_t)DM * 2048 * 2);
  float*    ycat   = (float*)alloc((size_t)BATCH * C2 * 64 * 4);
  float*    ca1    = (float*)alloc((size_t)BATCH * MIP * 64 * 4);
  float*    ahw    = (float*)alloc((size_t)BATCH * C2 * 64 * 4);
  _Float16* actp   = (_Float16*)alloc((size_t)BATCH * 2048 * 2);
  float*    mean1  = (float*)alloc(C1 * 4);
  float*    rstd1  = (float*)alloc(C1 * 4);
  float*    mean2  = (float*)alloc(C2 * 4);
  float*    rstd2  = (float*)alloc(C2 * 4);
  float*    mean3  = (float*)alloc(MIP * 4);
  float*    rstd3  = (float*)alloc(MIP * 4);

  cvt_w2_kernel<<<(9 * C2 * C1 + 255) / 256, 256, 0, stream>>>(conv2_w, w2t);
  cvt_f16_kernel<<<(DM * 2048 + 255) / 256, 256, 0, stream>>>(fc_w, fcwh, DM * 2048);
  zero_h_kernel<<<(NPAD + 255) / 256, 256, 0, stream>>>(x1p, NPAD);

  crop_resize_kernel<<<(BATCH * 3 * PR * PR) / 256, 256, 0, stream>>>(image, position, patch);

  conv1_kernel<<<(BATCH * C1 * PR * PR) / 256, 256, 0, stream>>>(patch, conv1_w, conv1_b, y1);
  stats_kernel<<<C1, 256, 0, stream>>>(y1, mean1, rstd1, BATCH, C1, PR * PR);
  bn_relu_pool_kernel<<<(BATCH * C1 * 1024) / 256, 256, 0, stream>>>(
      y1, mean1, rstd1, bn1_g, bn1_b, x1p);

  conv2_wmma_kernel<<<BATCH * 32, 512, 0, stream>>>(x1p, w2t, conv2_b, y2);
  stats_kernel<<<C2, 256, 0, stream>>>(y2, mean2, rstd2, BATCH, C2, 1024);
  bn2_relu_kernel<<<(BATCH * C2 * 1024) / 256, 256, 0, stream>>>(
      y2, mean2, rstd2, bn2_g, bn2_b, x2);

  ca_pool_kernel<<<(BATCH * C2 * 64) / 256, 256, 0, stream>>>(x2, ycat);
  ca1_kernel<<<(BATCH * MIP * 64) / 256, 256, 0, stream>>>(ycat, ca_c1_w, ca_c1_b, ca1);
  stats_kernel<<<MIP, 256, 0, stream>>>(ca1, mean3, rstd3, BATCH, MIP, 64);
  ca_bnrelu_kernel<<<(BATCH * MIP * 64) / 256, 256, 0, stream>>>(
      ca1, mean3, rstd3, ca_bn_g, ca_bn_b);
  ca_att_kernel<<<(BATCH * C2 * 64) / 256, 256, 0, stream>>>(
      ca1, ca_ch_w, ca_ch_b, ca_cw_w, ca_cw_b, ahw);

  pool16_kernel<<<(BATCH * C2 * 16) / 256, 256, 0, stream>>>(x2, ahw, actp);
  fc_wmma_kernel<<<32, 256, 0, stream>>>(actp, fcwh, fc_b, out);
}
